// ScaleFreeCrossAttention_64742337020421
// MI455X (gfx1250) — matrix operand order repaired and checked by reading, not compiled, not run on hardware
//
#include <hip/hip_runtime.h>
#include <hip/hip_bf16.h>
#include <math.h>

#define EMBED 768
#define HEADS 12
#define HDIM  64
#define FFDIM (EMBED * 2)
#define BATCH 4
#define LQ 2048
#define LK 2048

typedef __attribute__((ext_vector_type(16))) __bf16 v16bf;
typedef __attribute__((ext_vector_type(8)))  __bf16 v8bf;
typedef __attribute__((ext_vector_type(4)))  __bf16 v4bf;
typedef __attribute__((ext_vector_type(8)))  float  v8f;
typedef int v4i_vs __attribute__((vector_size(16)));   // matches builtin param type

// ---- CDNA5 async global->LDS path (ASYNCcnt), with safe fallback ----
#if __has_builtin(__builtin_amdgcn_global_load_async_to_lds_b128) && \
    __has_builtin(__builtin_amdgcn_s_wait_asynccnt)
#define USE_ASYNC_LDS 1
#else
#define USE_ASYNC_LDS 0
#endif

__device__ __forceinline__ void copy16_g2l(__bf16* ldst, const __bf16* gsrc) {
#if USE_ASYNC_LDS
    __builtin_amdgcn_global_load_async_to_lds_b128(
        (__attribute__((address_space(1))) v4i_vs*)gsrc,
        (__attribute__((address_space(3))) v4i_vs*)ldst, 0, 0);
#else
    *(v8bf*)ldst = *(const v8bf*)gsrc;
#endif
}
__device__ __forceinline__ void async_wait_all() {
#if USE_ASYNC_LDS
    __builtin_amdgcn_s_wait_asynccnt(0);
#endif
}

// ---- WMMA fragment layouts (CDNA5 ISA) ----
// A (16x32 bf16): lane holds M=lane%16; elem j -> K = (j<8?j:j+8) + (lane&16?8:0)
// B (32x16 bf16): lane holds N=lane%16; elem j -> K = (j<8?j:j+8) + (lane&16?8:0)
// C/D (16x16 f32): lane holds N=lane%16; vgpr e -> M = e + (lane&16?8:0)

__device__ __forceinline__ v16bf frag_a16(const __bf16* base, int off8) {
    v8bf lo = *(const v8bf*)(base + off8);
    v8bf hi = *(const v8bf*)(base + off8 + 16);
    return __builtin_shufflevector(lo, hi, 0,1,2,3,4,5,6,7,8,9,10,11,12,13,14,15);
}
__device__ __forceinline__ v16bf frag_b16(const __bf16* base, int off16) {
    v8bf lo = *(const v8bf*)(base + off16 / 2);
    v8bf hi = *(const v8bf*)(base + off16 / 2 + 16);
    return __builtin_shufflevector(lo, hi, 0,1,2,3,4,5,6,7,8,9,10,11,12,13,14,15);
}
__device__ __forceinline__ v4bf cvt4(float4 f) {
    v4bf r;
    r[0] = (__bf16)f.x; r[1] = (__bf16)f.y; r[2] = (__bf16)f.z; r[3] = (__bf16)f.w;
    return r;
}
__device__ __forceinline__ v8f wmma_bf16(v16bf a, v16bf b, v8f c) {
    return __builtin_amdgcn_wmma_f32_16x16x32_bf16(
        false, a, false, b, (short)0, c, false, false);
}

// =====================================================================
// one-time f32 -> bf16 conversion (vectorized)
// =====================================================================
__global__ __launch_bounds__(256)
void cvt_f32_bf16_kernel(const float* __restrict__ in, __bf16* __restrict__ out, int n4) {
    int i = blockIdx.x * 256 + threadIdx.x;
    if (i < n4) {
        float4 f = ((const float4*)in)[i];
        *(v4bf*)&out[(size_t)i * 4] = cvt4(f);
    }
}

// =====================================================================
// one-time weight transpose+convert: W[K][N] f32 -> Wt[N][K] bf16
// =====================================================================
__global__ __launch_bounds__(256)
void w_transpose_bf16_kernel(const float* __restrict__ W, __bf16* __restrict__ Wt,
                             int K, int N) {
    size_t idx = (size_t)blockIdx.x * 256 + threadIdx.x;   // over K*N, coalesced read
    int k = (int)(idx / N), n = (int)(idx % N);
    Wt[(size_t)n * K + k] = (__bf16)W[idx];
}

// =====================================================================
// GEMM: out[M,N] = epilogue( A[M,K](bf16) @ Wt[N,K](bf16)^T + bias (+res f32) )
// block tile 128x64, 8 waves of 32x32 (2x2 WMMA tiles), K-step 32.
// A double-buffered in LDS via async(128b) copies; B-frags straight from
// L2-resident transposed bf16 weights (2x global_load_b128 + prefetch).
// OUTF32: store f32, else bf16. TOUT: bf16 output stored transposed [N][M].
// =====================================================================
template <bool GELU, bool RES, bool OUTF32, bool TOUT>
__global__ __launch_bounds__(256)
void gemm_bf16_kernel(const __bf16* __restrict__ A, const __bf16* __restrict__ Wt,
                      const float* __restrict__ bias, const float* __restrict__ res,
                      void* __restrict__ Cout, int M, int N, int K) {
    __shared__ alignas(16) __bf16 As[2][128][32];   // double-buffered [m][k]

    const int tid  = threadIdx.x;
    const int lane = tid & 31;
    const int wave = tid >> 5;
    const int wr   = wave >> 1;   // 0..3
    const int wc   = wave & 1;    // 0..1

    const int rowBase = blockIdx.y * 128;
    const int colBase = blockIdx.x * 64;
    const int off8  = (lane & 16) ? 8 : 0;
    const int off16 = (lane & 16) ? 16 : 0;
    const int l15   = lane & 15;

    const __bf16* bptr[2];
    #pragma unroll
    for (int ni = 0; ni < 2; ++ni)
        bptr[ni] = Wt + (size_t)(colBase + wc * 32 + ni * 16 + l15) * K;

    // staging: 512 v8bf = 128*32/8, 2 per thread
    const int sr0 = tid >> 2,         sc0 = (tid & 3) * 8;
    const int sr1 = (tid + 256) >> 2, sc1 = sc0;
    auto stage = [&](int buf, int k0) {
        copy16_g2l(&As[buf][sr0][sc0], &A[(size_t)(rowBase + sr0) * K + k0 + sc0]);
        copy16_g2l(&As[buf][sr1][sc1], &A[(size_t)(rowBase + sr1) * K + k0 + sc1]);
    };

    v8f acc[2][2] = {};
    stage(0, 0);
    int cur = 0;

    for (int k0 = 0; k0 < K; k0 += 32) {
        async_wait_all();
        __syncthreads();
        if (k0 + 32 < K) {
            stage(cur ^ 1, k0 + 32);                       // overlap with WMMAs
            __builtin_prefetch(bptr[0] + k0 + 32, 0, 3);   // next B-frag rows
            __builtin_prefetch(bptr[1] + k0 + 32, 0, 3);
        }

        v16bf afrag[2], bfrag[2];
        #pragma unroll
        for (int mi = 0; mi < 2; ++mi)
            afrag[mi] = frag_a16(&As[cur][wr * 32 + mi * 16 + l15][0], off8);
        #pragma unroll
        for (int ni = 0; ni < 2; ++ni)
            bfrag[ni] = frag_b16(bptr[ni] + k0, off16);
        #pragma unroll
        for (int mi = 0; mi < 2; ++mi)
            #pragma unroll
            for (int ni = 0; ni < 2; ++ni)
                acc[mi][ni] = wmma_bf16(afrag[mi], bfrag[ni], acc[mi][ni]);
        __syncthreads();
        cur ^= 1;
    }

    const int mhalf = (lane & 16) ? 8 : 0;
    #pragma unroll
    for (int mi = 0; mi < 2; ++mi) {
        #pragma unroll
        for (int ni = 0; ni < 2; ++ni) {
            int n = colBase + wc * 32 + ni * 16 + l15;
            float bval = bias[n];
            #pragma unroll
            for (int e = 0; e < 8; ++e) {
                int m = rowBase + wr * 32 + mi * 16 + e + mhalf;
                float val = acc[mi][ni][e] + bval;
                if (RES) val += res[(size_t)m * N + n];
                if (GELU) val = 0.5f * val * (1.0f + erff(val * 0.70710678118654752f));
                if (OUTF32) {
                    ((float*)Cout)[(size_t)m * N + n] = val;
                } else {
                    __bf16* Cb = (__bf16*)Cout;
                    Cb[TOUT ? ((size_t)n * M + m) : ((size_t)m * N + n)] = (__bf16)val;
                }
            }
        }
    }
}

// =====================================================================
// Per-head L2 normalize in place on bf16: x viewed as [B*L*H, 64]
// =====================================================================
__global__ __launch_bounds__(256)
void l2norm_heads_kernel(__bf16* __restrict__ x, int total) {
    int widx = blockIdx.x * 8 + (threadIdx.x >> 5);
    int lane = threadIdx.x & 31;
    if (widx >= total) return;
    __bf16* p = x + (size_t)widx * 64;
    float a = (float)p[lane];
    float b = (float)p[lane + 32];
    float ss = a * a + b * b;
    #pragma unroll
    for (int o = 16; o > 0; o >>= 1) ss += __shfl_xor(ss, o, 32);
    float inv = 1.0f / fmaxf(sqrtf(ss), 1e-12f);
    p[lane]      = (__bf16)(a * inv);
    p[lane + 32] = (__bf16)(b * inv);
}

// =====================================================================
// Attention on bf16 q,k,vT. Scores in [-1,1] (l2-normed) => streaming
// softmax without max tracking. One block per (bh, 128 q rows); each
// wave owns 16 q rows. K staged row-major, V already transposed in
// global ([n][m]) so both LDS tiles are pure async 128-bit copies.
// =====================================================================
__global__ __launch_bounds__(256)
void attention_kernel(const __bf16* __restrict__ q, const __bf16* __restrict__ k,
                      const __bf16* __restrict__ vT, __bf16* __restrict__ out) {
    __shared__ alignas(16) __bf16 Ks[32][64];        // [key][d]
    __shared__ alignas(16) __bf16 VsT[64][32];       // [d][key]
    __shared__ alignas(16) __bf16 Pst[8][16][32];    // per-wave [m][kcol]

    const int tid  = threadIdx.x;
    const int lane = tid & 31;
    const int wave = tid >> 5;
    const int bh   = blockIdx.y;
    const int b    = bh / HEADS;
    const int h    = bh % HEADS;
    const int qbase = blockIdx.x * 128 + wave * 16;
    const int off8  = (lane & 16) ? 8 : 0;
    const int off16 = (lane & 16) ? 16 : 0;
    const int l15   = lane & 15;
    const int mhalf = (lane & 16) ? 8 : 0;
    const int M = BATCH * LQ;

    // Q fragments: contiguous bf16 runs -> 128-bit global loads
    v16bf qfrag[2];
    {
        const __bf16* qp = q + ((size_t)(b * LQ + qbase + l15)) * EMBED + h * HDIM;
        qfrag[0] = frag_a16(qp, off8);
        qfrag[1] = frag_a16(qp + 32, off8);
    }

    const int kr = tid >> 3, kc8 = (tid & 7) * 8;    // K tile: 256 v8bf
    const int vr = tid >> 2, vc8 = (tid & 3) * 8;    // V tile: 256 v8bf

    v8f num[4] = {};          // 16 x 64 accumulator (4 d-subtiles)
    float den[8] = {};        // per-lane partial softmax denominator

    for (int kk = 0; kk < LK; kk += 32) {
        copy16_g2l(&Ks[kr][kc8],
                   &k[((size_t)(b * LK + kk + kr)) * EMBED + h * HDIM + kc8]);
        copy16_g2l(&VsT[vr][vc8],
                   &vT[(size_t)(h * HDIM + vr) * M + b * LK + kk + vc8]);
        async_wait_all();
        __syncthreads();

        // scores: two 16x16 key tiles, each = sum of 2 d-chunk WMMAs
        v8f s[2] = {};
        #pragma unroll
        for (int ni = 0; ni < 2; ++ni)
            #pragma unroll
            for (int dc = 0; dc < 2; ++dc)
                s[ni] = wmma_bf16(qfrag[dc],
                                  frag_b16(&Ks[ni * 16 + l15][dc * 32], off16),
                                  s[ni]);

        // exp, accumulate denominator, stage P for the PV A-fragment
        #pragma unroll
        for (int ni = 0; ni < 2; ++ni) {
            int kcol = ni * 16 + l15;
            #pragma unroll
            for (int e = 0; e < 8; ++e) {
                float pval = __expf(s[ni][e]);   // s in [-1,1]: no max needed
                den[e] += pval;
                Pst[wave][e + mhalf][kcol] = (__bf16)pval;
            }
        }

        // P (16x32) @ V (32x64)
        v16bf pa = frag_a16(&Pst[wave][l15][0], off8);
        #pragma unroll
        for (int ni = 0; ni < 4; ++ni)
            num[ni] = wmma_bf16(pa, frag_b16(&VsT[ni * 16 + l15][0], off16), num[ni]);
        __syncthreads();
    }

    // reduce denominator across the 16 lanes of each half
    #pragma unroll
    for (int e = 0; e < 8; ++e) {
        float d = den[e];
        #pragma unroll
        for (int o = 8; o > 0; o >>= 1) d += __shfl_xor(d, o, 32);
        den[e] = d;
    }

    #pragma unroll
    for (int ni = 0; ni < 4; ++ni) {
        int d = ni * 16 + l15;
        #pragma unroll
        for (int e = 0; e < 8; ++e) {
            int row = qbase + e + mhalf;
            out[((size_t)(b * LQ + row)) * EMBED + h * HDIM + d] =
                (__bf16)(num[ni][e] / den[e]);
        }
    }
}

// =====================================================================
// out = LayerNorm(x + y) * g + beta ; optional extra bf16 copy of out
// =====================================================================
__global__ __launch_bounds__(256)
void add_ln_kernel(const float* __restrict__ x, const float* __restrict__ y,
                   const float* __restrict__ g, const float* __restrict__ beta,
                   float* __restrict__ out, __bf16* __restrict__ out_bf) {
    const int N = EMBED;
    const int row = blockIdx.x;
    const int tid = threadIdx.x;
    __shared__ float red[256];

    const float* xr = x + (size_t)row * N;
    const float* yr = y + (size_t)row * N;
    float v0 = xr[tid] + yr[tid];
    float v1 = xr[tid + 256] + yr[tid + 256];
    float v2 = xr[tid + 512] + yr[tid + 512];

    red[tid] = v0 + v1 + v2;
    __syncthreads();
    for (int o = 128; o > 0; o >>= 1) {
        if (tid < o) red[tid] += red[tid + o];
        __syncthreads();
    }
    float mu = red[0] * (1.0f / N);
    __syncthreads();

    float d0 = v0 - mu, d1 = v1 - mu, d2 = v2 - mu;
    red[tid] = d0 * d0 + d1 * d1 + d2 * d2;
    __syncthreads();
    for (int o = 128; o > 0; o >>= 1) {
        if (tid < o) red[tid] += red[tid + o];
        __syncthreads();
    }
    float rstd = rsqrtf(red[0] * (1.0f / N) + 1e-5f);

    float o0 = d0 * rstd * g[tid]       + beta[tid];
    float o1 = d1 * rstd * g[tid + 256] + beta[tid + 256];
    float o2 = d2 * rstd * g[tid + 512] + beta[tid + 512];
    float* orow = out + (size_t)row * N;
    orow[tid] = o0; orow[tid + 256] = o1; orow[tid + 512] = o2;
    if (out_bf) {
        __bf16* brow = out_bf + (size_t)row * N;
        brow[tid] = (__bf16)o0; brow[tid + 256] = (__bf16)o1; brow[tid + 512] = (__bf16)o2;
    }
}

// =====================================================================
extern "C" void kernel_launch(void* const* d_in, const int* in_sizes, int n_in,
                              void* d_out, int out_size, void* d_ws, size_t ws_size,
                              hipStream_t stream) {
    const float* image_tokens = (const float*)d_in[0];
    const float* point_tokens = (const float*)d_in[1];
    const float* image_pos    = (const float*)d_in[2];
    const float* point_pos    = (const float*)d_in[3];
    const float* Wip = (const float*)d_in[4];   const float* bip = (const float*)d_in[5];
    const float* Wpp = (const float*)d_in[6];   const float* bpp = (const float*)d_in[7];
    const float* Wq  = (const float*)d_in[8];   const float* bq  = (const float*)d_in[9];
    const float* Wk  = (const float*)d_in[10];  const float* bk  = (const float*)d_in[11];
    const float* Wv  = (const float*)d_in[12];  const float* bv  = (const float*)d_in[13];
    const float* Wo  = (const float*)d_in[14];  const float* bo  = (const float*)d_in[15];
    const float* g1  = (const float*)d_in[16];  const float* b1  = (const float*)d_in[17];
    const float* g2  = (const float*)d_in[18];  const float* b2  = (const float*)d_in[19];
    const float* Wf1 = (const float*)d_in[20];  const float* bf1 = (const float*)d_in[21];
    const float* Wf2 = (const float*)d_in[22];  const float* bf2 = (const float*)d_in[23];

    const int M = BATCH * LQ;                 // 8192
    const size_t SZ = (size_t)M * EMBED;      // 6.29M elements
    const size_t WSZ = (size_t)EMBED * EMBED;
    const size_t FSZ = (size_t)EMBED * FFDIM;

    char* w = (char*)d_ws;
    auto alloc = [&](size_t bytes) { char* p = w; w += (bytes + 255) & ~(size_t)255; return p; };

    __bf16* imgpos_bf = (__bf16*)alloc(SZ * 2);
    __bf16* ptpos_bf  = (__bf16*)alloc(SZ * 2);
    __bf16* ptok_bf   = (__bf16*)alloc(SZ * 2);
    __bf16* tmp1_bf   = (__bf16*)alloc(SZ * 2);
    __bf16* tmp2_bf   = (__bf16*)alloc(SZ * 2);
    __bf16* qb        = (__bf16*)alloc(SZ * 2);
    __bf16* kb        = (__bf16*)alloc(SZ * 2);
    __bf16* vT        = (__bf16*)alloc(SZ * 2);           // [EMBED][M]
    __bf16* attn_bf   = (__bf16*)alloc(SZ * 2);
    __bf16* fused_bf  = (__bf16*)alloc(SZ * 2);
    __bf16* hbuf_bf   = (__bf16*)alloc((size_t)M * FFDIM * 2);
    float*  proj      = (float*)alloc(SZ * 4);
    float*  fused     = (float*)alloc(SZ * 4);
    float*  ffb       = (float*)alloc(SZ * 4);
    __bf16* WipT = (__bf16*)alloc(WSZ * 2);
    __bf16* WppT = (__bf16*)alloc(WSZ * 2);
    __bf16* WqT  = (__bf16*)alloc(WSZ * 2);
    __bf16* WkT  = (__bf16*)alloc(WSZ * 2);
    __bf16* WvT  = (__bf16*)alloc(WSZ * 2);
    __bf16* WoT  = (__bf16*)alloc(WSZ * 2);
    __bf16* Wf1T = (__bf16*)alloc(FSZ * 2);
    __bf16* Wf2T = (__bf16*)alloc(FSZ * 2);

    const dim3 blk(256);
    const dim3 g768(EMBED / 64, M / 128);
    const dim3 g1536(FFDIM / 64, M / 128);

    // one-time conversions
    {
        int n4 = (int)(SZ / 4), blocks = (n4 + 255) / 256;
        cvt_f32_bf16_kernel<<<blocks, blk, 0, stream>>>(image_pos, imgpos_bf, n4);
        cvt_f32_bf16_kernel<<<blocks, blk, 0, stream>>>(point_pos, ptpos_bf, n4);
        cvt_f32_bf16_kernel<<<blocks, blk, 0, stream>>>(point_tokens, ptok_bf, n4);
        int wb = (int)(WSZ / 256), fb = (int)(FSZ / 256);
        w_transpose_bf16_kernel<<<wb, blk, 0, stream>>>(Wip, WipT, EMBED, EMBED);
        w_transpose_bf16_kernel<<<wb, blk, 0, stream>>>(Wpp, WppT, EMBED, EMBED);
        w_transpose_bf16_kernel<<<wb, blk, 0, stream>>>(Wq,  WqT,  EMBED, EMBED);
        w_transpose_bf16_kernel<<<wb, blk, 0, stream>>>(Wk,  WkT,  EMBED, EMBED);
        w_transpose_bf16_kernel<<<wb, blk, 0, stream>>>(Wv,  WvT,  EMBED, EMBED);
        w_transpose_bf16_kernel<<<wb, blk, 0, stream>>>(Wo,  WoT,  EMBED, EMBED);
        w_transpose_bf16_kernel<<<fb, blk, 0, stream>>>(Wf1, Wf1T, EMBED, FFDIM);
        w_transpose_bf16_kernel<<<fb, blk, 0, stream>>>(Wf2, Wf2T, FFDIM, EMBED);
    }

    // pos-enhanced inputs (residual f32 tokens, bf16 out)
    gemm_bf16_kernel<false, true, false, false><<<g768, blk, 0, stream>>>(imgpos_bf, WipT, bip, image_tokens, tmp1_bf, M, EMBED, EMBED);
    gemm_bf16_kernel<false, true, false, false><<<g768, blk, 0, stream>>>(ptpos_bf, WppT, bpp, point_tokens, tmp2_bf, M, EMBED, EMBED);
    // q/k/v projections (v written transposed [N][M] for attention staging)
    gemm_bf16_kernel<false, false, false, false><<<g768, blk, 0, stream>>>(tmp1_bf, WqT, bq, nullptr, qb, M, EMBED, EMBED);
    gemm_bf16_kernel<false, false, false, false><<<g768, blk, 0, stream>>>(tmp2_bf, WkT, bk, nullptr, kb, M, EMBED, EMBED);
    gemm_bf16_kernel<false, false, false, true ><<<g768, blk, 0, stream>>>(ptok_bf, WvT, bv, nullptr, vT, M, EMBED, EMBED);
    // per-head l2 normalization
    {
        int total = M * HEADS;
        int blocks = (total + 7) / 8;
        l2norm_heads_kernel<<<blocks, blk, 0, stream>>>(qb, total);
        l2norm_heads_kernel<<<blocks, blk, 0, stream>>>(kb, total);
    }
    // fused attention
    {
        dim3 ga(LQ / 128, BATCH * HEADS);
        attention_kernel<<<ga, blk, 0, stream>>>(qb, kb, vT, attn_bf);
    }
    // output projection + residual layernorm (fused kept f32 + bf16 copy)
    gemm_bf16_kernel<false, false, true, false><<<g768, blk, 0, stream>>>(attn_bf, WoT, bo, nullptr, proj, M, EMBED, EMBED);
    add_ln_kernel<<<M, blk, 0, stream>>>(image_tokens, proj, g1, b1, fused, fused_bf);
    // FFN with exact GELU
    gemm_bf16_kernel<true,  false, false, false><<<g1536, blk, 0, stream>>>(fused_bf, Wf1T, bf1, nullptr, hbuf_bf, M, FFDIM, EMBED);
    gemm_bf16_kernel<false, false, true, false><<<g768,  blk, 0, stream>>>(hbuf_bf, Wf2T, bf2, nullptr, ffb, M, EMBED, FFDIM);
    add_ln_kernel<<<M, blk, 0, stream>>>(fused, ffb, g2, b2, (float*)d_out, nullptr);
}
